// MultiHeadAttention_36971078484252
// MI455X (gfx1250) — compile-verified, matrix operand order checked by reading, not run on hardware
//
#include <hip/hip_runtime.h>
#include <hip/hip_bf16.h>
#include <stdint.h>

#define DEV __device__ __forceinline__
#define AS1 __attribute__((address_space(1)))
#define AS3 __attribute__((address_space(3)))

#if defined(__HIP_DEVICE_COMPILE__) && __has_builtin(__builtin_amdgcn_global_load_async_to_lds_b128)
#define HAVE_ASYNC_LDS 1
#else
#define HAVE_ASYNC_LDS 0
#endif

typedef _Float16 half_t;
typedef __attribute__((ext_vector_type(16))) _Float16 v16h;
typedef __attribute__((ext_vector_type(8)))  _Float16 v8h;
typedef __attribute__((ext_vector_type(4)))  _Float16 v4h;
typedef __attribute__((ext_vector_type(8)))  float    v8f;
typedef __attribute__((ext_vector_type(4)))  float    v4f;
typedef __attribute__((ext_vector_type(4)))  int      v4i;

// Problem constants (fixed by the reference)
constexpr int Bb = 4, Ss = 1024, Dd = 2048, Hh = 16, HDd = 128;
constexpr float INV_SCALE = 0.08838834764831845f; // 1/sqrt(128)

// Tiling
constexpr int BM = 128, BN = 128, BK = 32;
constexpr int LDT = BK + 8;   // 40 halves = 80B row stride (multiple of 16B)

enum { EP_PLAIN = 0, EP_QKV = 1, EP_CTX = 2 };

DEV void wait_async0() {
#if defined(__HIP_DEVICE_COMPILE__)
#if __has_builtin(__builtin_amdgcn_s_wait_asynccnt)
  __builtin_amdgcn_s_wait_asynccnt(0);
#else
  asm volatile("s_wait_asynccnt 0x0" ::: "memory");
#endif
#endif
}

// build a 16x32 f16 fragment from LDS per ISA 7.12.2 layout
DEV v16h frag_from_lds(const half_t* __restrict__ S, int row, int kb) {
  v8h lo = *(const v8h*)(S + row * LDT + kb);       // K = kb .. kb+7
  v8h hi = *(const v8h*)(S + row * LDT + kb + 16);  // K = kb+16 .. kb+23
  v16h a;
#pragma unroll
  for (int i = 0; i < 8; ++i) { a[i] = lo[i]; a[i + 8] = hi[i]; }
  return a;
}

// ---- generic WMMA GEMM: Out = scale * (A @ op(W)) + bias --------------------
// TRANSB=true : C[m,n] = sum_k A[m,k] * W[n,k]   (Linear with W stored [N,K])
// TRANSB=false: C[m,n] = sum_k A[m,k] * W[k,n]   (P @ V)
// Software-pipelined, LDS ping-pong; f16 operands use async global->LDS DMA.
template <typename AT, typename WT, typename OT, int EPI, bool HASBIAS, bool TRANSB>
__global__ __launch_bounds__(256) void wmma_gemm(
    const AT* __restrict__ A, const WT* __restrict__ W,
    const float* __restrict__ bias, OT* __restrict__ Out,
    int M, int N, int K,
    long strideAz, long strideWz, long strideOz,
    int ldOut, float scale) {
  __shared__ __attribute__((aligned(16))) half_t As[2][BM * LDT];
  __shared__ __attribute__((aligned(16))) half_t Bs[2][BN * LDT];

  const int tid  = threadIdx.x;
  const int wave = tid >> 5;
  const int lid  = tid & 31;
  const int hl   = lid & 15;   // lane within half-wave
  const int hi   = lid >> 4;   // 0/1 -> K base 0/8
  const int wm   = wave & 1;   // M offset *64
  const int wn   = wave >> 1;  // N offset *32

  const int Mb = blockIdx.y * BM;
  const int Nb = blockIdx.x * BN;
  const int z  = blockIdx.z;

  const AT* Az = A + (long)z * strideAz;
  const WT* Wz = W + (long)z * strideWz;

  constexpr bool AF32 = (sizeof(AT) == 4);
  constexpr bool WF32 = (sizeof(WT) == 4);
  constexpr int ANCH = AF32 ? 4 : 2;
  constexpr int WNCH = (TRANSB && WF32) ? 4 : 2;
  // async DMA path usable only for f16 row-major tile copies (no convert/transpose)
  constexpr bool ASY_A = (HAVE_ASYNC_LDS != 0) && !AF32;
  constexpr bool ASY_W = (HAVE_ASYNC_LDS != 0) && TRANSB && !WF32;

  // ---- per-thread fetch pointers + LDS store offsets (computed once) ----
  const AT* aP[ANCH];
  int aOff[ANCH];
#pragma unroll
  for (int it = 0; it < ANCH; ++it) {
    const int idx = tid + it * 256;
    int row, k;
    if constexpr (AF32) { row = idx >> 3; k = (idx & 7) * 4; }   // 4 floats/chunk
    else                { row = idx >> 2; k = (idx & 3) * 8; }   // 8 halves/chunk
    aP[it]  = Az + (long)(Mb + row) * K + k;
    aOff[it] = row * LDT + k;
  }
  const WT* wP[WNCH];
  int wOff[WNCH];
#pragma unroll
  for (int it = 0; it < WNCH; ++it) {
    const int idx = tid + it * 256;
    if constexpr (TRANSB) {
      int row, k;
      if constexpr (WF32) { row = idx >> 3; k = (idx & 7) * 4; }
      else                { row = idx >> 2; k = (idx & 3) * 8; }
      wP[it]  = Wz + (long)(Nb + row) * K + k;
      wOff[it] = row * LDT + k;
    } else {  // V [K,N] row-major, N == 128: coalesced row fetch, transpose on LDS store
      const int k = idx >> 4;           // 16 v8h chunks per 128-wide row
      const int n = (idx & 15) * 8;
      wP[it]  = Wz + (long)k * N + n;
      wOff[it] = idx;                   // decode (k,n) again at store time
    }
  }

  // ---- staging registers + fetch/store/DMA helpers ----
  v4f aF4[4]; v8h aF8[2];
  v4f wF4[4]; v8h wF8[2];

  auto fetchA = [&] {
    if constexpr (AF32) {
#pragma unroll
      for (int it = 0; it < 4; ++it) { aF4[it] = *(const v4f*)aP[it]; aP[it] += BK; }
    } else {
#pragma unroll
      for (int it = 0; it < 2; ++it) { aF8[it] = *(const v8h*)aP[it]; aP[it] += BK; }
    }
  };
  auto fetchW = [&] {
    if constexpr (TRANSB) {
      if constexpr (WF32) {
#pragma unroll
        for (int it = 0; it < 4; ++it) { wF4[it] = *(const v4f*)wP[it]; wP[it] += BK; }
      } else {
#pragma unroll
        for (int it = 0; it < 2; ++it) { wF8[it] = *(const v8h*)wP[it]; wP[it] += BK; }
      }
    } else {
#pragma unroll
      for (int it = 0; it < 2; ++it) { wF8[it] = *(const v8h*)wP[it]; wP[it] += (long)BK * N; }
    }
  };
  auto storeA = [&](half_t* __restrict__ S) {
    if constexpr (AF32) {
#pragma unroll
      for (int it = 0; it < 4; ++it) {
        v4h vh;
        vh[0] = (half_t)aF4[it][0]; vh[1] = (half_t)aF4[it][1];
        vh[2] = (half_t)aF4[it][2]; vh[3] = (half_t)aF4[it][3];
        *(v4h*)(S + aOff[it]) = vh;
      }
    } else {
#pragma unroll
      for (int it = 0; it < 2; ++it) *(v8h*)(S + aOff[it]) = aF8[it];
    }
  };
  auto storeW = [&](half_t* __restrict__ S) {
    if constexpr (TRANSB) {
      if constexpr (WF32) {
#pragma unroll
        for (int it = 0; it < 4; ++it) {
          v4h vh;
          vh[0] = (half_t)wF4[it][0]; vh[1] = (half_t)wF4[it][1];
          vh[2] = (half_t)wF4[it][2]; vh[3] = (half_t)wF4[it][3];
          *(v4h*)(S + wOff[it]) = vh;
        }
      } else {
#pragma unroll
        for (int it = 0; it < 2; ++it) *(v8h*)(S + wOff[it]) = wF8[it];
      }
    } else {  // scatter-transpose: S[n][k]
#pragma unroll
      for (int it = 0; it < 2; ++it) {
        const int idx = wOff[it];
        const int k = idx >> 4;
        const int n = (idx & 15) * 8;
#pragma unroll
        for (int i = 0; i < 8; ++i) S[(n + i) * LDT + k] = wF8[it][i];
      }
    }
  };
  // async DMA: 16B/lane straight global -> LDS (ASYNCcnt)
  auto copyA = [&](half_t* __restrict__ S) {
#if HAVE_ASYNC_LDS
    if constexpr (ASY_A) {
#pragma unroll
      for (int it = 0; it < 2; ++it) {
        __builtin_amdgcn_global_load_async_to_lds_b128(
            (AS1 v4i*)(uintptr_t)aP[it],
            (AS3 v4i*)(uint32_t)(uintptr_t)(S + aOff[it]), 0, 0);
        aP[it] += BK;
      }
    }
#else
    (void)S;
#endif
  };
  auto copyW = [&](half_t* __restrict__ S) {
#if HAVE_ASYNC_LDS
    if constexpr (ASY_W) {
#pragma unroll
      for (int it = 0; it < 2; ++it) {
        __builtin_amdgcn_global_load_async_to_lds_b128(
            (AS1 v4i*)(uintptr_t)wP[it],
            (AS3 v4i*)(uint32_t)(uintptr_t)(S + wOff[it]), 0, 0);
        wP[it] += BK;
      }
    }
#else
    (void)S;
#endif
  };

  // ---- accumulators ----
  const v8f zero = {0.f, 0.f, 0.f, 0.f, 0.f, 0.f, 0.f, 0.f};
  v8f acc[4][2];
#pragma unroll
  for (int i = 0; i < 4; ++i)
#pragma unroll
    for (int j = 0; j < 2; ++j) acc[i][j] = zero;

  // ---- prologue: tile 0 into buffer 0 ----
  if constexpr (ASY_A) copyA(As[0]); else fetchA();
  if constexpr (ASY_W) copyW(Bs[0]); else fetchW();
  if constexpr (!ASY_A) storeA(As[0]);
  if constexpr (!ASY_W) storeW(Bs[0]);
  if constexpr (ASY_A || ASY_W) wait_async0();
  __syncthreads();

  half_t* Ap = As[0]; half_t* Aq = As[1];
  half_t* Bp = Bs[0]; half_t* Bq = Bs[1];

  const int kSteps = K / BK;
  for (int ks = 0; ks < kSteps; ++ks) {
    const bool hasNext = (ks + 1) < kSteps;
    if (hasNext) {
      // issue next tile's data movement; it flies under the WMMAs below.
      // Aq/Bq were last read at ks-1, before the barrier ending ks-1 -> safe.
      if constexpr (ASY_A) copyA(Aq); else fetchA();
      if constexpr (ASY_W) copyW(Bq); else fetchW();
      if constexpr (AF32) __builtin_prefetch(aP[0], 0, 1);  // global_prefetch_b8
      if constexpr (TRANSB && WF32) __builtin_prefetch(wP[0], 0, 1);
    }

    v16h afr[4], bfr[2];
#pragma unroll
    for (int mt = 0; mt < 4; ++mt)
      afr[mt] = frag_from_lds(Ap, wm * 64 + mt * 16 + hl, hi * 8);
#pragma unroll
    for (int nt = 0; nt < 2; ++nt)
      bfr[nt] = frag_from_lds(Bp, wn * 32 + nt * 16 + hl, hi * 8);

#pragma unroll
    for (int mt = 0; mt < 4; ++mt)
#pragma unroll
      for (int nt = 0; nt < 2; ++nt)
        acc[mt][nt] = __builtin_amdgcn_wmma_f32_16x16x32_f16(
            false, afr[mt], false, bfr[nt], (short)0, acc[mt][nt], false, false);

    if (hasNext) {
      if constexpr (!ASY_A) storeA(Aq);          // staged loadcnt wait here
      if constexpr (!ASY_W) storeW(Bq);
      if constexpr (ASY_A || ASY_W) wait_async0();
      __syncthreads();                           // tile ks+1 visible to all
      half_t* t;
      t = Ap; Ap = Aq; Aq = t;
      t = Bp; Bp = Bq; Bq = t;
    }
  }

  // ---- epilogue ----
#pragma unroll
  for (int mt = 0; mt < 4; ++mt) {
#pragma unroll
    for (int nt = 0; nt < 2; ++nt) {
      const int nl = wn * 32 + nt * 16 + hl;
      const int n  = Nb + nl;
      float bv = 0.f;
      if constexpr (HASBIAS) bv = bias[n];
#pragma unroll
      for (int r = 0; r < 8; ++r) {
        const int ml = wm * 64 + mt * 16 + r + hi * 8;
        const int m  = Mb + ml;
        const float val = acc[mt][nt][r] * scale + bv;
        if constexpr (EPI == EP_PLAIN) {
          Out[(long)z * strideOz + (long)m * ldOut + n] = (OT)val;
        } else if constexpr (EPI == EP_QKV) {
          // m = b*S+s over [4096], n = h*HD+hd over [2048] -> [B,H,S,HD] f16
          const int b = m >> 10, s = m & 1023;
          const int h = n >> 7,  hd = n & 127;
          Out[(((long)b * Hh + h) * Ss + s) * HDd + hd] = (OT)val;
        } else { // EP_CTX: z = b*H+h, m = s, n = hd -> ctx [B,S,D] f16
          const int b = z >> 4, h = z & 15;
          Out[((long)b * Ss + m) * Dd + h * HDd + n] = (OT)val;
        }
      }
    }
  }
}

// ---- row softmax over 1024-wide rows, in place ------------------------------
__global__ __launch_bounds__(256) void softmax_rows(float* __restrict__ attn) {
  __shared__ float red[256];
  const long row = blockIdx.x;
  float* p = attn + row * 1024;
  const int tid = threadIdx.x;

  v4f x = *(const v4f*)(p + tid * 4);
  float m = fmaxf(fmaxf(x[0], x[1]), fmaxf(x[2], x[3]));
  red[tid] = m;
  __syncthreads();
#pragma unroll
  for (int s = 128; s > 0; s >>= 1) {
    if (tid < s) red[tid] = fmaxf(red[tid], red[tid + s]);
    __syncthreads();
  }
  m = red[0];
  __syncthreads();

  v4f e;
  float sum = 0.f;
#pragma unroll
  for (int i = 0; i < 4; ++i) { e[i] = __expf(x[i] - m); sum += e[i]; }
  red[tid] = sum;
  __syncthreads();
#pragma unroll
  for (int s = 128; s > 0; s >>= 1) {
    if (tid < s) red[tid] += red[tid + s];
    __syncthreads();
  }
  const float inv = 1.0f / red[0];
  e *= inv;
  *(v4f*)(p + tid * 4) = e;
}

extern "C" void kernel_launch(void* const* d_in, const int* in_sizes, int n_in,
                              void* d_out, int out_size, void* d_ws, size_t ws_size,
                              hipStream_t stream) {
  (void)in_sizes; (void)n_in; (void)out_size; (void)ws_size;
  const float* q    = (const float*)d_in[0];
  const float* k    = (const float*)d_in[1];
  const float* v    = (const float*)d_in[2];
  const float* wq_w = (const float*)d_in[3];
  const float* wq_b = (const float*)d_in[4];
  const float* wk_w = (const float*)d_in[5];
  const float* wk_b = (const float*)d_in[6];
  const float* wv_w = (const float*)d_in[7];
  const float* wv_b = (const float*)d_in[8];
  const float* wo_w = (const float*)d_in[9];
  const float* wo_b = (const float*)d_in[10];

  float* out  = (float*)d_out;                 // [B,S,D]
  float* attn = out + (long)Bb * Ss * Dd;      // [B,H,S,S]

  const long headElems = (long)Bb * Hh * Ss * HDd; // 8.4M halves each
  half_t* Qh = (half_t*)d_ws;
  half_t* Kh = Qh + headElems;
  half_t* Vh = Kh + headElems;
  half_t* Ch = Qh;                             // reuse Qh after scores (ctx f16 [B,S,D])

  dim3 blk(256);
  dim3 gProj(Dd / BN, (Bb * Ss) / BM, 1);      // (16,32,1)
  dim3 gSc(Ss / BN, Ss / BM, Bb * Hh);         // (8,8,64)
  dim3 gCtx(HDd / BN, Ss / BM, Bb * Hh);       // (1,8,64)

  // Q/K/V projections: f32 in, f16 head-split out
  wmma_gemm<float, float, half_t, EP_QKV, true, true><<<gProj, blk, 0, stream>>>(
      q, wq_w, wq_b, Qh, Bb * Ss, Dd, Dd, 0, 0, 0, 0, 1.0f);
  wmma_gemm<float, float, half_t, EP_QKV, true, true><<<gProj, blk, 0, stream>>>(
      k, wk_w, wk_b, Kh, Bb * Ss, Dd, Dd, 0, 0, 0, 0, 1.0f);
  wmma_gemm<float, float, half_t, EP_QKV, true, true><<<gProj, blk, 0, stream>>>(
      v, wv_w, wv_b, Vh, Bb * Ss, Dd, Dd, 0, 0, 0, 0, 1.0f);

  // scores = Q @ K^T / sqrt(HD), batched over 64 heads, f32 into attn output
  wmma_gemm<half_t, half_t, float, EP_PLAIN, false, true><<<gSc, blk, 0, stream>>>(
      Qh, Kh, nullptr, attn, Ss, Ss, HDd,
      (long)Ss * HDd, (long)Ss * HDd, (long)Ss * Ss, Ss, INV_SCALE);

  // softmax over last dim, in place
  softmax_rows<<<dim3(Bb * Hh * Ss), blk, 0, stream>>>(attn);

  // context = P @ V -> ctx f16 [B,S,D]
  wmma_gemm<float, half_t, half_t, EP_CTX, false, false><<<gCtx, blk, 0, stream>>>(
      attn, Vh, nullptr, Ch, Ss, HDd, Ss,
      (long)Ss * Ss, (long)Ss * HDd, 0, 0, 1.0f);

  // output projection: ctx f16 @ Wo^T + bo -> f32 [B,S,D]
  wmma_gemm<half_t, float, float, EP_PLAIN, true, true><<<gProj, blk, 0, stream>>>(
      Ch, wo_w, wo_b, out, Bb * Ss, Dd, Dd, 0, 0, 0, Dd, 1.0f);
}
